// AngularPenaltySMLoss_1056561955366
// MI455X (gfx1250) — matrix operand order repaired and checked by reading, not compiled, not run on hardware
//
#include <hip/hip_runtime.h>
#include <hip/hip_bf16.h>

// ---------------- CDNA5 WMMA types ----------------
typedef __attribute__((ext_vector_type(16))) __bf16 v16bf;
typedef __attribute__((ext_vector_type(8)))  float  v8f;

union Frag { v16bf v; uint4 u[2]; };

#define B_ROWS   1024
#define D_DIM    512
#define C_CLS    100000
#define BN       128            // classes per block
#define BM       64             // rows per m-chunk (4 M-waves x 16)
#define LDS_STR  520            // padded bf16 row stride (conflict-free b128)
#define N_CT     ((C_CLS + BN - 1) / BN)   // 782 class tiles

// ---------------------------------------------------------------------------
// 1) L2-normalize features rows -> bf16 xn in workspace. One wave per row.
// ---------------------------------------------------------------------------
__global__ __launch_bounds__(256)
void normalize_kernel(const float* __restrict__ f, __bf16* __restrict__ xn)
{
    const int lane = threadIdx.x & 31;
    const int wave = threadIdx.x >> 5;
    const int row  = blockIdx.x * 8 + wave;

    const float* fr = f + (size_t)row * D_DIM + lane * 16;
    float4 v[4];
    float ss = 0.0f;
#pragma unroll
    for (int i = 0; i < 4; ++i) {
        v[i] = *(const float4*)(fr + i * 4);
        ss += v[i].x * v[i].x + v[i].y * v[i].y + v[i].z * v[i].z + v[i].w * v[i].w;
    }
#pragma unroll
    for (int m = 1; m < 32; m <<= 1) ss += __shfl_xor(ss, m, 32);

    const float inv = 1.0f / fmaxf(sqrtf(ss), 1e-12f);

    union { __bf16 h[16]; uint4 u[2]; } o;
#pragma unroll
    for (int i = 0; i < 4; ++i) {
        o.h[4 * i + 0] = (__bf16)(v[i].x * inv);
        o.h[4 * i + 1] = (__bf16)(v[i].y * inv);
        o.h[4 * i + 2] = (__bf16)(v[i].z * inv);
        o.h[4 * i + 3] = (__bf16)(v[i].w * inv);
    }
    __bf16* out = xn + (size_t)row * D_DIM + lane * 16;
    *(uint4*)out       = o.u[0];
    *((uint4*)out + 1) = o.u[1];
}

// ---------------------------------------------------------------------------
// 2) True-class logit tgt[i] = dot(features[i], W[y[i]]) / ||features[i]||
//    in full f32 (margin math is precision sensitive). One wave per row.
// ---------------------------------------------------------------------------
__global__ __launch_bounds__(256)
void target_kernel(const float* __restrict__ f, const float* __restrict__ W,
                   const int* __restrict__ y, float* __restrict__ tgt)
{
    const int lane = threadIdx.x & 31;
    const int wave = threadIdx.x >> 5;
    const int row  = blockIdx.x * 8 + wave;
    const int cls  = y[row];

    const float* fr = f + (size_t)row * D_DIM;
    const float* wr = W + (size_t)cls * D_DIM;
    float dot = 0.0f, ss = 0.0f;
#pragma unroll 4
    for (int j = lane; j < D_DIM; j += 32) {
        float a = fr[j];
        dot += a * wr[j];
        ss  += a * a;
    }
#pragma unroll
    for (int m = 1; m < 32; m <<= 1) {
        dot += __shfl_xor(dot, m, 32);
        ss  += __shfl_xor(ss,  m, 32);
    }
    if (lane == 0) tgt[row] = dot / fmaxf(sqrtf(ss), 1e-12f);
}

// ---------------------------------------------------------------------------
// 3) Main fused kernel: per block, stage a 128-class W slice (f32->bf16) into
//    LDS once (130 KB, CDNA5 320KB/WGP), then loop over all 1024 rows doing
//    bf16 WMMA GEMM fused with exp(S*logit), class masking, and per-row
//    partial reduction.
//    8 waves = 4 (M) x 2 (N); per wave 16x64 output = 4 WMMA accumulators
//    (4 independent v_wmma per K-step, 2.5 b128 loads per WMMA).
//    Deterministic: every partials[ct][row] slot has exactly one writer.
// ---------------------------------------------------------------------------
__global__ __launch_bounds__(256, 1)
void arcface_gemm_exp_kernel(const __bf16* __restrict__ xn,
                             const float*  __restrict__ W,
                             float*        __restrict__ partials)
{
    extern __shared__ char smem[];
    __bf16* wt     = (__bf16*)smem;                      // BN x LDS_STR bf16
    float*  rowacc = (float*)(smem + BN * LDS_STR * 2);  // [BM][2] per-chunk

    const int tid  = threadIdx.x;
    const int lane = tid & 31;
    const int wave = tid >> 5;
    const int mw   = wave >> 1;          // 0..3  (M)
    const int nw   = wave & 1;           // 0..1  (N, 64 classes each)
    const int lo   = lane & 15;
    const int hi   = lane >> 4;
    const int cbase = blockIdx.x * BN;

    // ---- stage W tile: 128 x 512 f32 -> bf16 LDS (zeros for class >= C) ----
#pragma unroll 4
    for (int i = 0; i < (BN * D_DIM / 4) / 256; ++i) {   // 64 float4 per thread
        int fid = tid + 256 * i;
        int row = fid >> 7;              // / (512/4)
        int c4  = fid & 127;
        float4 v = make_float4(0.f, 0.f, 0.f, 0.f);
        int cls = cbase + row;
        if (cls < C_CLS)
            v = *(const float4*)(W + (size_t)cls * D_DIM + c4 * 4);
        union { __bf16 h[4]; uint2 u; } pk;
        pk.h[0] = (__bf16)v.x; pk.h[1] = (__bf16)v.y;
        pk.h[2] = (__bf16)v.z; pk.h[3] = (__bf16)v.w;
        *(uint2*)(wt + row * LDS_STR + c4 * 4) = pk.u;
    }
    __syncthreads();

    // column-validity masks per N-fragment (same for every row chunk)
    bool vf[4];
    const __bf16* brow[4];
#pragma unroll
    for (int f = 0; f < 4; ++f) {
        vf[f]   = (cbase + nw * 64 + f * 16 + lo) < C_CLS;
        brow[f] = wt + (nw * 64 + f * 16 + lo) * LDS_STR + hi * 8;
    }

    for (int mc = 0; mc < B_ROWS / BM; ++mc) {
        const int m0 = mc * BM + mw * 16;
        v8f c[4];
#pragma unroll
        for (int f = 0; f < 4; ++f) c[f] = (v8f){0.f,0.f,0.f,0.f,0.f,0.f,0.f,0.f};

        // A fragment base: row = m0+lo, K chunks at kk + hi*8 and +16
        const __bf16* arow = xn + (size_t)(m0 + lo) * D_DIM + hi * 8;

#pragma unroll 2
        for (int kk = 0; kk < D_DIM; kk += 32) {
            Frag a;
            a.u[0] = *(const uint4*)(arow + kk);
            a.u[1] = *(const uint4*)(arow + kk + 16);
            Frag b[4];
#pragma unroll
            for (int f = 0; f < 4; ++f) {
                b[f].u[0] = *(const uint4*)(brow[f] + kk);
                b[f].u[1] = *(const uint4*)(brow[f] + kk + 16);
            }
#pragma unroll
            for (int f = 0; f < 4; ++f) {
                c[f] = __builtin_amdgcn_wmma_f32_16x16x32_bf16(
                           false, a.v, false, b[f].v, (short)0, c[f], false, false);
            }
        }

        // exp(S * logit) with class masking, combine the four N-frags
        float et[8];
#pragma unroll
        for (int j = 0; j < 8; ++j) {
            float e = 0.0f;
#pragma unroll
            for (int f = 0; f < 4; ++f)
                e += vf[f] ? __expf(64.0f * c[f][j]) : 0.0f;
            et[j] = e;
        }
        // butterfly over the 16-lane halves: every lane in a half ends with
        // the half's column-sum; VGPR j <-> row j (lanes 0-15) / j+8 (16-31)
#pragma unroll
        for (int m = 1; m < 16; m <<= 1) {
#pragma unroll
            for (int j = 0; j < 8; ++j) et[j] += __shfl_xor(et[j], m, 32);
        }
        if (lane == 0) {
#pragma unroll
            for (int j = 0; j < 8; ++j) rowacc[(mw * 16 + j) * 2 + nw] = et[j];
        } else if (lane == 16) {
#pragma unroll
            for (int j = 0; j < 8; ++j) rowacc[(mw * 16 + 8 + j) * 2 + nw] = et[j];
        }
        __syncthreads();
        if (tid < BM) {
            const float* r = rowacc + tid * 2;
            partials[(size_t)blockIdx.x * B_ROWS + mc * BM + tid] = r[0] + r[1];
        }
        __syncthreads();
    }
}

// ---------------------------------------------------------------------------
// 4) Final loss: per row combine 782 partials, apply arcface margin, reduce.
//    cos(acos(t)+M) = t*cosM - sqrt(1-t^2)*sinM  (avoids acos/cos).
// ---------------------------------------------------------------------------
__global__ __launch_bounds__(1024)
void loss_kernel(const float* __restrict__ tgt, const float* __restrict__ partials,
                 float* __restrict__ out)
{
    __shared__ float red[1024];
    const int i = threadIdx.x;

    float t = tgt[i];
    t = fminf(fmaxf(t, -1.0f + 1e-7f), 1.0f - 1e-7f);
    const float cM = 0.87758256189037276f;   // cos(0.5)
    const float sM = 0.47942553860420301f;   // sin(0.5)
    float num = 64.0f * (t * cM - sqrtf(1.0f - t * t) * sM);

    float full = 0.0f;
    for (int ct = 0; ct < N_CT; ++ct)
        full += partials[(size_t)ct * B_ROWS + i];      // coalesced

    float denom = __expf(num) + (full - __expf(64.0f * t));
    float L = num - __logf(denom);
    red[i] = -L * (1.0f / (float)B_ROWS);
    __syncthreads();
#pragma unroll
    for (int s = 512; s > 0; s >>= 1) {
        if (i < s) red[i] += red[i + s];
        __syncthreads();
    }
    if (i == 0) out[0] = red[0];
}

// ---------------------------------------------------------------------------
extern "C" void kernel_launch(void* const* d_in, const int* in_sizes, int n_in,
                              void* d_out, int out_size, void* d_ws, size_t ws_size,
                              hipStream_t stream)
{
    const float* features = (const float*)d_in[0];   // [1024,512] f32
    const float* W        = (const float*)d_in[1];   // [100000,512] f32
    const int*   y        = (const int*)d_in[2];     // [1024]
    float*       out      = (float*)d_out;           // scalar f32

    // workspace layout (~4.1 MB total, all 16B-aligned)
    char*   ws       = (char*)d_ws;
    __bf16* xn       = (__bf16*)ws;                               // 1 MB
    float*  tgt      = (float*)(ws + (size_t)B_ROWS * D_DIM * 2); // 4 KB
    float*  partials = (float*)(ws + (size_t)B_ROWS * D_DIM * 2 + 4096); // 3.2 MB

    normalize_kernel<<<B_ROWS / 8, 256, 0, stream>>>(features, xn);
    target_kernel  <<<B_ROWS / 8, 256, 0, stream>>>(features, W, y, tgt);

    const size_t lds_bytes = (size_t)BN * LDS_STR * 2 + BM * 2 * sizeof(float);
    arcface_gemm_exp_kernel<<<N_CT, 256, lds_bytes, stream>>>(xn, W, partials);

    loss_kernel<<<1, 1024, 0, stream>>>(tgt, partials, out);
}
